// PhysicsEngine_41351945126389
// MI455X (gfx1250) — compile-verified, matrix operand order checked by reading, not run on hardware
//
#include <hip/hip_runtime.h>
#include <math.h>

// ---------------- constants ----------------
#define Nn   5000
#define MN   5008          // Nn padded to multiple of 16
#define Ee   100000        // already multiple of 16
#define G2   256
#define R2C  (0.145f*0.145f)

typedef __attribute__((ext_vector_type(16))) _Float16 v16h;
typedef __attribute__((ext_vector_type(8)))  float    v8f;

// ---------------- device helpers ----------------
__device__ __forceinline__ float gelu_f(float x){
  return 0.5f*x*(1.f + tanhf(0.7978845608028654f*(x + 0.044715f*x*x*x)));
}
__device__ __forceinline__ float lin16(int i){ return -1.f + (2.f/15.f)*(float)i; }

// Load 16 f32 at p[0..7], p[16..23] and convert to the f16 A/B fragment layout
// (CDNA5 ISA 7.12.2: lane<16 holds K{0..7,16..23}, lane>=16 holds K{8..15,24..31};
//  caller passes p already offset by half*8).
__device__ __forceinline__ v16h mk_a(const float* p){
  float4 a = *(const float4*)(p);
  float4 b = *(const float4*)(p+4);
  float4 c = *(const float4*)(p+16);
  float4 d = *(const float4*)(p+20);
  v16h r;
  r[0]=(_Float16)a.x;  r[1]=(_Float16)a.y;  r[2]=(_Float16)a.z;  r[3]=(_Float16)a.w;
  r[4]=(_Float16)b.x;  r[5]=(_Float16)b.y;  r[6]=(_Float16)b.z;  r[7]=(_Float16)b.w;
  r[8]=(_Float16)c.x;  r[9]=(_Float16)c.y;  r[10]=(_Float16)c.z; r[11]=(_Float16)c.w;
  r[12]=(_Float16)d.x; r[13]=(_Float16)d.y; r[14]=(_Float16)d.z; r[15]=(_Float16)d.w;
  return r;
}

// ---------------- generic WMMA GEMM with async-LDS weight staging ----------------
// C[M,dN] = act(A[M,Kp] x Wt^T + bias), Wt stored [NT*16, Kp] (pre-transposed).
// The 32-wide K-slice of the weight panel (NT*16 x 32 f32) is staged into LDS by the
// whole block via global_load_async_to_lds_b128 (ASYNCcnt), double-buffered.
// act: 0 none, 1 relu, 2 gelu.  Stores guarded by (Mreal, dNreal).
template<int NT>
__global__ __launch_bounds__(256)
void k_gemm(const float* __restrict__ A, const float* __restrict__ Wt,
            const float* __restrict__ bias, float* __restrict__ C,
            int M, int Mreal, int Kp, int dNreal, int ldc, int act)
{
  __shared__ float bp[2][NT*512];               // NT*16 rows x 32 cols, x2 buffers
  int wave = threadIdx.x >> 5, lane = threadIdx.x & 31;
  int mt = blockIdx.x * 8 + wave;
  bool act_w = (mt * 16 < M);                   // wave-uniform; inactive waves still stage
  int half = lane >> 4, lr = lane & 15;
  const float* ap = A + (long)(mt*16 + lr) * Kp;
  v8f acc[NT] = {};
  int KT = Kp >> 5;
  const int CH = NT * 128;                      // b128 chunks per 32-wide slice

  // stage slice 0 into buffer 0
  for (int c = threadIdx.x; c < CH; c += 256){
    int row = c >> 3, q = (c & 7) << 2;
    const float* g = Wt + (long)row*Kp + q;
    unsigned l = (unsigned)(unsigned long long)(const void*)&bp[0][row*32 + q];
    asm volatile("global_load_async_to_lds_b128 %0, %1, off" :: "v"(l), "v"(g) : "memory");
  }
  int buf = 0;
  for (int kt = 0; kt < KT; ++kt){
    asm volatile("s_wait_asynccnt 0x0" ::: "memory");
    __syncthreads();                            // all copies landed; other buffer free
    if (kt + 1 < KT){
      for (int c = threadIdx.x; c < CH; c += 256){
        int row = c >> 3, q = (c & 7) << 2;
        const float* g = Wt + (long)row*Kp + (kt+1)*32 + q;
        unsigned l = (unsigned)(unsigned long long)(const void*)&bp[buf^1][row*32 + q];
        asm volatile("global_load_async_to_lds_b128 %0, %1, off" :: "v"(l), "v"(g) : "memory");
      }
    }
    if (act_w){
      int k0 = kt*32 + half*8;
      __builtin_prefetch(ap + k0 + 64, 0, 1);
      v16h av = mk_a(ap + k0);
#pragma unroll
      for (int nt = 0; nt < NT; ++nt){
        v16h bv = mk_a(&bp[buf][(nt*16 + lr)*32 + half*8]);
        acc[nt] = __builtin_amdgcn_wmma_f32_16x16x32_f16(false, av, false, bv,
                                                         (short)0, acc[nt], false, false);
      }
    }
    buf ^= 1;
  }
  if (!act_w) return;
#pragma unroll
  for (int nt = 0; nt < NT; ++nt){
    int n = nt*16 + lr;
    float b = bias ? bias[n] : 0.f;
#pragma unroll
    for (int j = 0; j < 8; ++j){
      int m = mt*16 + j + half*8;
      float v = acc[nt][j] + b;
      if (act == 1) v = fmaxf(v, 0.f);
      else if (act == 2) v = gelu_f(v);
      if (m < Mreal && n < dNreal) C[(long)m*ldc + n] = v;
    }
  }
}

// ---------------- layernorm over width-128 rows ----------------
__global__ __launch_bounds__(256)
void k_layernorm(const float* __restrict__ in, float* __restrict__ out, int rows)
{
  int row = blockIdx.x*8 + (threadIdx.x >> 5);
  if (row >= rows) return;
  int lane = threadIdx.x & 31;
  const float* p = in + (long)row*128;
  float v0=p[lane], v1=p[lane+32], v2=p[lane+64], v3=p[lane+96];
  float s = v0+v1+v2+v3;
  for (int m=16; m>=1; m>>=1) s += __shfl_xor(s, m, 32);
  float mu = s * (1.f/128.f);
  float d0=v0-mu, d1=v1-mu, d2=v2-mu, d3=v3-mu;
  float q = d0*d0+d1*d1+d2*d2+d3*d3;
  for (int m=16; m>=1; m>>=1) q += __shfl_xor(q, m, 32);
  float r = rsqrtf(q*(1.f/128.f) + 1e-5f);
  float* o = out + (long)row*128;
  o[lane]=d0*r; o[lane+32]=d1*r; o[lane+64]=d2*r; o[lane+96]=d3*r;
}

// ---------------- small utility kernels ----------------
__global__ void k_zero(float* __restrict__ p, long n){
  long i = (long)blockIdx.x*blockDim.x + threadIdx.x;
  long st = (long)gridDim.x*blockDim.x;
  for (; i < n; i += st) p[i] = 0.f;
}
__global__ void k_add(float* __restrict__ d, const float* __restrict__ s, long n){
  long i = (long)blockIdx.x*blockDim.x + threadIdx.x;
  long st = (long)gridDim.x*blockDim.x;
  for (; i < n; i += st) d[i] += s[i];
}
// Wt[n,k] = W[(rowOff+k)*Nout + n], zero-padded to [NoutPad, Kp]
__global__ void k_xposew(const float* __restrict__ w, float* __restrict__ wt,
                         int rowOff, int Krows, int Nout, int NoutPad, int Kp)
{
  int id = blockIdx.x*256 + threadIdx.x;
  if (id >= NoutPad*Kp) return;
  int nn = id / Kp, k = id % Kp;
  float v = 0.f;
  if (nn < Nout && k < Krows) v = w[(long)(rowOff + k)*Nout + nn];
  wt[id] = v;
}
__global__ void k_node_in_build(const int* __restrict__ x, const float* __restrict__ pos,
                                const float* __restrict__ emb, float* __restrict__ X0)
{
  int n = blockIdx.x, t = threadIdx.x;          // grid MN, block 64
  float v = 0.f;
  if (n < Nn){
    if (t < 16) v = emb[x[n]*16 + t];
    else if (t < 37) v = pos[(long)n*21 + (t-16)];
  }
  X0[(long)n*64 + t] = v;
}
__global__ void k_edge_in_build(const float* __restrict__ ea, float* __restrict__ E0)
{
  int i = blockIdx.x, t = threadIdx.x;          // grid Ee, block 32
  E0[(long)i*32 + t] = (t < 4) ? ea[(long)i*4 + t] : 0.f;
}
__global__ void k_edge_concat(const float* __restrict__ h, const float* __restrict__ e,
                              const int* __restrict__ src, const int* __restrict__ dst,
                              float* __restrict__ eC)
{
  int i = blockIdx.x, t = threadIdx.x;          // grid Ee, block 128
  eC[(long)i*384       + t] = h[(long)src[i]*128 + t];
  eC[(long)i*384 + 128 + t] = h[(long)dst[i]*128 + t];
  eC[(long)i*384 + 256 + t] = e[(long)i*128 + t];
}
__global__ void k_scale_rows(float* __restrict__ m, const float* __restrict__ nd)
{
  int i = blockIdx.x, t = threadIdx.x;          // grid Ee, block 128
  m[(long)i*128 + t] *= nd[i];
}
__global__ void k_scatter_add(const float* __restrict__ m, const int* __restrict__ dst,
                              float* __restrict__ agg)
{
  int i = blockIdx.x, t = threadIdx.x;          // grid Ee, block 128
  unsafeAtomicAdd(&agg[(long)dst[i]*128 + t], m[(long)i*128 + t]);
}
__global__ void k_node_concat(const float* __restrict__ h, const float* __restrict__ agg,
                              float* __restrict__ nC)
{
  int n = blockIdx.x, t = threadIdx.x;          // grid MN, block 128
  nC[(long)n*256 + t]       = h[(long)n*128 + t];
  nC[(long)n*256 + 128 + t] = agg[(long)n*128 + t];
}
__global__ void k_project2d(const float* __restrict__ rp, const float* __restrict__ w,
                            const float* __restrict__ b, float* __restrict__ p2)
{
  int n = blockIdx.x*blockDim.x + threadIdx.x;
  if (n >= Nn) return;
  float a0=rp[n*3], a1=rp[n*3+1], a2=rp[n*3+2];
  p2[n*2+0] = tanhf(a0*w[0] + a1*w[2] + a2*w[4] + b[0]);
  p2[n*2+1] = tanhf(a0*w[1] + a1*w[3] + a2*w[5] + b[1]);
}
__global__ void k_qe(float* __restrict__ qe)
{
  int m = threadIdx.x;                          // block 256
  float c0 = lin16(m & 15), c1 = lin16(m >> 4); // meshgrid 'xy': (lin[j], lin[i])
  for (int k = 0; k < 16; ++k){
    float f = (float)(1 << k);
    qe[m*64 + k]      = sinf(c0*f);
    qe[m*64 + 16 + k] = cosf(c0*f);
    qe[m*64 + 32 + k] = sinf(c1*f);
    qe[m*64 + 48 + k] = cosf(c1*f);
  }
}
__global__ void k_yf(const float* __restrict__ p2, const float* __restrict__ out16,
                     float* __restrict__ yf)
{
  int n = blockIdx.x, t = threadIdx.x;          // grid Nn, block 64
  float c = p2[n*2 + (t >> 5)];
  int u = t & 31, k = u & 15;
  float ang = c * (float)(1 << k);
  yf[(long)n*96 + t] = (u < 16) ? sinf(ang) : cosf(ang);
  if (t < 3) yf[(long)n*96 + 64 + t] = out16[n*16 + t];
}
__global__ void k_w_in(const float* __restrict__ p2, float* __restrict__ w_in)
{
  __shared__ float red[256];
  __shared__ float inv_s;
  int m = blockIdx.x, t = threadIdx.x;          // grid 256, block 256
  float qx = lin16(m & 15), qy = lin16(m >> 4);
  float cnt = 0.f;
  for (int n = t; n < Nn; n += 256){
    float dx = qx - p2[n*2], dy = qy - p2[n*2+1];
    cnt += (dx*dx + dy*dy <= R2C) ? 1.f : 0.f;
  }
  red[t] = cnt; __syncthreads();
  for (int s = 128; s > 0; s >>= 1){ if (t < s) red[t] += red[t+s]; __syncthreads(); }
  if (t == 0) inv_s = 1.f / fmaxf(red[0], 1.f);
  __syncthreads();
  float inv = inv_s;
  for (int n = t; n < MN; n += 256){
    float wv = 0.f;
    if (n < Nn){
      float dx = qx - p2[n*2], dy = qy - p2[n*2+1];
      if (dx*dx + dy*dy <= R2C) wv = inv;
    }
    w_in[(long)m*MN + n] = wv;
  }
}
__global__ void k_w_out(const float* __restrict__ p2, float* __restrict__ w_out)
{
  __shared__ float red[256];
  __shared__ float inv_s;
  int n = blockIdx.x, g = threadIdx.x;          // grid Nn, block 256
  float dx = lin16(g & 15) - p2[n*2], dy = lin16(g >> 4) - p2[n*2+1];
  float msk = (dx*dx + dy*dy <= R2C) ? 1.f : 0.f;
  red[g] = msk; __syncthreads();
  for (int s = 128; s > 0; s >>= 1){ if (g < s) red[g] += red[g+s]; __syncthreads(); }
  if (g == 0) inv_s = 1.f / fmaxf(red[0], 1.f);
  __syncthreads();
  w_out[(long)n*256 + g] = msk * inv_s;
}

// ---------------- FNO (direct 16-point DFTs, tiny) ----------------
__global__ void k_rfft2(const float* __restrict__ u, float* __restrict__ ur, float* __restrict__ ui)
{
  int id = blockIdx.x*128 + threadIdx.x;        // 16*9*32 = 4608
  if (id >= 4608) return;
  int c = id & 31, t = id >> 5, ky = t % 9, kx = t / 9;
  float sr = 0.f, si = 0.f;
  for (int a = 0; a < 16; ++a)
    for (int b = 0; b < 16; ++b){
      float ang = -6.2831853071795864f * (float)(kx*a + ky*b) * (1.f/16.f);
      float v = u[(long)(a*16 + b)*32 + c];
      sr += v * cosf(ang);
      si += v * sinf(ang);
    }
  ur[id] = sr; ui[id] = si;
}
__global__ void k_specmul(const float* __restrict__ ur, const float* __restrict__ ui,
                          const float* __restrict__ wr, const float* __restrict__ wi,
                          float* __restrict__ outr, float* __restrict__ outi)
{
  int id = blockIdx.x*128 + threadIdx.x;        // 4608
  if (id >= 4608) return;
  int d = id & 31, t = id >> 5;                 // t = kx*9+ky
  float ar = 0.f, ai = 0.f;
  for (int c = 0; c < 32; ++c){
    float xr = ur[t*32 + c], xi = ui[t*32 + c];
    long wb = ((long)t*32 + c)*32 + d;
    float wrv = wr[wb], wiv = wi[wb];
    ar += xr*wrv - xi*wiv;
    ai += xr*wiv + xi*wrv;
  }
  outr[id] = ar; outi[id] = ai;
}
__global__ void k_irfft2(const float* __restrict__ outr, const float* __restrict__ outi,
                         float* __restrict__ spec)
{
  int id = blockIdx.x*128 + threadIdx.x;        // 256*32 = 8192
  if (id >= 8192) return;
  int c = id & 31, g = id >> 5, x = g >> 4, y = g & 15;
  float s = 0.f;
  for (int kx = 0; kx < 16; ++kx)
    for (int ky = 0; ky < 9; ++ky){
      float ang = 6.2831853071795864f * (float)(kx*x + ky*y) * (1.f/16.f);
      float wgt = (ky == 0 || ky == 8) ? 1.f : 2.f;
      int idx = (kx*9 + ky)*32 + c;
      s += wgt * (outr[idx]*cosf(ang) - outi[idx]*sinf(ang));
    }
  spec[(long)g*32 + c] = s * (1.f/256.f);
}
__global__ void k_gelu_add(float* __restrict__ u, const float* __restrict__ spec,
                           const float* __restrict__ us)
{
  int i = blockIdx.x*128 + threadIdx.x;         // 8192
  if (i >= 8192) return;
  u[i] = gelu_f(spec[i] + us[i]);
}

// ---------------- fused GINO pairwise kernels (WMMA) ----------------
// in-GNO: latentP[m, 0:3] += sum_n w_in[m,n] * K3( gelu(qA[m]+B1[n]) )
__global__ __launch_bounds__(128)
void k_gino_in(const float* __restrict__ qA, const float* __restrict__ B1,
               const float* __restrict__ Wt2, const float* __restrict__ b2,
               const float* __restrict__ Wt3, const float* __restrict__ b3,
               const float* __restrict__ w_in, float* __restrict__ latentP,
               int tilesPerWave)
{
  __shared__ float lds[4][1024];                // 16x64 f32 per wave
  int wave = threadIdx.x >> 5, lane = threadIdx.x & 31;
  int half = lane >> 4, lr = lane & 15;
  int m = blockIdx.x;
  int chunk = blockIdx.y*4 + wave;
  int k0 = half*8;
  float qa[16];
  { const float* qp = qA + m*32 + k0;
    #pragma unroll
    for (int i = 0; i < 8; ++i){ qa[i] = qp[i]; qa[i+8] = qp[i+16]; } }
  v16h bv2[4];
  #pragma unroll
  for (int nt = 0; nt < 4; ++nt) bv2[nt] = mk_a(Wt2 + (long)(nt*16 + lr)*32 + k0);
  v16h bv3[2];
  #pragma unroll
  for (int kt = 0; kt < 2; ++kt) bv3[kt] = mk_a(Wt3 + (long)lr*64 + kt*32 + k0);
  float b2f[4];
  #pragma unroll
  for (int nt = 0; nt < 4; ++nt) b2f[nt] = b2[nt*16 + lr];

  float part = 0.f, wpart = 0.f;
  for (int t = 0; t < tilesPerWave; ++t){
    int nb = (chunk*tilesPerWave + t)*16;
    if (nb >= MN) break;
    const float* bp = B1 + (long)(nb + lr)*32 + k0;
    v16h av;
    #pragma unroll
    for (int i = 0; i < 8; ++i){
      av[i]   = (_Float16)gelu_f(qa[i]   + bp[i]);
      av[i+8] = (_Float16)gelu_f(qa[i+8] + bp[i+16]);
    }
    v8f acc[4] = {};
    #pragma unroll
    for (int nt = 0; nt < 4; ++nt)
      acc[nt] = __builtin_amdgcn_wmma_f32_16x16x32_f16(false, av, false, bv2[nt],
                                                       (short)0, acc[nt], false, false);
    #pragma unroll
    for (int nt = 0; nt < 4; ++nt)
      #pragma unroll
      for (int j = 0; j < 8; ++j)
        lds[wave][(j + half*8)*64 + nt*16 + lr] = gelu_f(acc[nt][j] + b2f[nt]);
    asm volatile("s_wait_dscnt 0x0" ::: "memory");   // LDS store -> cross-lane load
    v8f k3 = {};
    #pragma unroll
    for (int kt = 0; kt < 2; ++kt){
      v16h a2 = mk_a(&lds[wave][lr*64 + kt*32 + k0]);
      k3 = __builtin_amdgcn_wmma_f32_16x16x32_f16(false, a2, false, bv3[kt],
                                                  (short)0, k3, false, false);
    }
    const float* wp = w_in + (long)m*MN + nb;
    #pragma unroll
    for (int j = 0; j < 8; ++j){
      float wv = wp[j + half*8];
      part  += k3[j]*wv;
      wpart += wv;
    }
  }
  part  += __shfl_xor(part, 16, 32);
  wpart += __shfl_xor(wpart, 16, 32);
  if (lane < 3) unsafeAtomicAdd(&latentP[m*32 + lane], part + b3[lane]*wpart);
}

// out-GNO: z[n,c] = sum_g w_out[n,g]*u[g,c]*(K3(gelu(qA2[n]+B2[g]))[g,c]+b3[c])
__global__ __launch_bounds__(128)
void k_gino_out(const float* __restrict__ qA2, const float* __restrict__ B2,
                const float* __restrict__ Wt2, const float* __restrict__ b2,
                const float* __restrict__ Wt3, const float* __restrict__ b3,
                const float* __restrict__ w_out, const float* __restrict__ u,
                float* __restrict__ z)
{
  __shared__ float lds[4][1024];
  int wave = threadIdx.x >> 5, lane = threadIdx.x & 31;
  int half = lane >> 4, lr = lane & 15;
  int n = blockIdx.x*4 + wave;
  if (n >= Nn) return;
  int k0 = half*8;
  float qa[16];
  { const float* qp = qA2 + (long)n*32 + k0;
    #pragma unroll
    for (int i = 0; i < 8; ++i){ qa[i] = qp[i]; qa[i+8] = qp[i+16]; } }
  v16h bv2[4];
  #pragma unroll
  for (int nt = 0; nt < 4; ++nt) bv2[nt] = mk_a(Wt2 + (long)(nt*16 + lr)*32 + k0);
  v16h bv3[2][2];
  #pragma unroll
  for (int nt = 0; nt < 2; ++nt)
    #pragma unroll
    for (int kt = 0; kt < 2; ++kt)
      bv3[nt][kt] = mk_a(Wt3 + (long)(nt*16 + lr)*64 + kt*32 + k0);
  float b2f[4];
  #pragma unroll
  for (int nt = 0; nt < 4; ++nt) b2f[nt] = b2[nt*16 + lr];

  float p0=0.f, q0=0.f, p1=0.f, q1=0.f;
  for (int gt = 0; gt < 16; ++gt){
    int gb = gt*16;
    const float* bp = B2 + (long)(gb + lr)*32 + k0;
    v16h av;
    #pragma unroll
    for (int i = 0; i < 8; ++i){
      av[i]   = (_Float16)gelu_f(qa[i]   + bp[i]);
      av[i+8] = (_Float16)gelu_f(qa[i+8] + bp[i+16]);
    }
    v8f acc[4] = {};
    #pragma unroll
    for (int nt = 0; nt < 4; ++nt)
      acc[nt] = __builtin_amdgcn_wmma_f32_16x16x32_f16(false, av, false, bv2[nt],
                                                       (short)0, acc[nt], false, false);
    #pragma unroll
    for (int nt = 0; nt < 4; ++nt)
      #pragma unroll
      for (int j = 0; j < 8; ++j)
        lds[wave][(j + half*8)*64 + nt*16 + lr] = gelu_f(acc[nt][j] + b2f[nt]);
    asm volatile("s_wait_dscnt 0x0" ::: "memory");
    v8f k3a = {}, k3b = {};
    #pragma unroll
    for (int kt = 0; kt < 2; ++kt){
      v16h a2 = mk_a(&lds[wave][lr*64 + kt*32 + k0]);
      k3a = __builtin_amdgcn_wmma_f32_16x16x32_f16(false, a2, false, bv3[0][kt],
                                                   (short)0, k3a, false, false);
      k3b = __builtin_amdgcn_wmma_f32_16x16x32_f16(false, a2, false, bv3[1][kt],
                                                   (short)0, k3b, false, false);
    }
    const float* wp = w_out + (long)n*256 + gb;
    #pragma unroll
    for (int j = 0; j < 8; ++j){
      int g = gb + j + half*8;
      float wv = wp[j + half*8];
      float f0 = wv * u[(long)g*32 + lr];
      float f1 = wv * u[(long)g*32 + 16 + lr];
      p0 += k3a[j]*f0; q0 += f0;
      p1 += k3b[j]*f1; q1 += f1;
    }
  }
  p0 += __shfl_xor(p0, 16, 32); q0 += __shfl_xor(q0, 16, 32);
  p1 += __shfl_xor(p1, 16, 32); q1 += __shfl_xor(q1, 16, 32);
  if (lane < 16){
    z[(long)n*32 + lr]      = p0 + b3[lr]*q0;
    z[(long)n*32 + 16 + lr] = p1 + b3[16 + lr]*q1;
  }
}

// ---------------- host orchestration ----------------
static void gemm_go(hipStream_t s, const float* A, const float* Wt, const float* bias,
                    float* C, int M, int Mreal, int Kp, int dN, int dNreal, int ldc, int act)
{
  int blocks = (M/16 + 7)/8;
  if (dN <= 16)
    k_gemm<1><<<blocks, 256, 0, s>>>(A, Wt, bias, C, M, Mreal, Kp, dNreal, ldc, act);
  else if (dN <= 32)
    k_gemm<2><<<blocks, 256, 0, s>>>(A, Wt, bias, C, M, Mreal, Kp, dNreal, ldc, act);
  else
    k_gemm<8><<<blocks, 256, 0, s>>>(A, Wt, bias, C, M, Mreal, Kp, dNreal, ldc, act);
}

extern "C" void kernel_launch(void* const* d_in, const int* in_sizes, int n_in,
                              void* d_out, int out_size, void* d_ws, size_t ws_size,
                              hipStream_t stream)
{
  if (n_in < 127) return;   // expects 6 data inputs + 121 param leaves (jax pytree order)
  const int*   xv  = (const int*)  d_in[0];
  const float* pos = (const float*)d_in[1];
  const float* ea  = (const float*)d_in[2];
  const int*   ei  = (const int*)  d_in[3];
  const float* nd  = (const float*)d_in[4];
  const float* rp  = (const float*)d_in[5];
  const float* P[127];
  for (int i = 0; i < 127; ++i) P[i] = (const float*)d_in[i];
  const int* src = ei, *dst = ei + Ee;

  // leaf index map (dicts sorted, lists in order, each linear = {b,w}):
  // 6:edge_in(6) 12:embed 13:fno0{skip.b,skip.w,wi,wr} 17:fno1 21:in_kernel(6)
  // 27:lifting(4) 31:out_kernel(6) 37:projection(4) 41:in_layers(36)
  // 77:node_in(6) 83:node_out(6) 89:out_layers(36) 125:project2d{b,w}

  float* base = (float*)d_ws;
  size_t off = 0;
  auto alloc = [&](size_t n)->float*{ float* p = base + off; off += (n + 63) & ~(size_t)63; return p; };
  auto prep = [&](const float* w, int rowOff, int Krows, int Nout, int NoutPad, int Kp)->float*{
    float* wt = alloc((size_t)NoutPad*Kp);
    int tot = NoutPad*Kp;
    k_xposew<<<(tot + 255)/256, 256, 0, stream>>>(w, wt, rowOff, Krows, Nout, NoutPad, Kp);
    return wt;
  };

  // ---- pre-transpose/pad all weights ----
  float* niW[3] = { prep(P[78], 0, 37, 128, 128, 64),
                    prep(P[80], 0, 128, 128, 128, 128),
                    prep(P[82], 0, 128, 128, 128, 128) };
  const float* niB[3] = { P[77], P[79], P[81] };
  float* eiW[3] = { prep(P[7], 0, 4, 128, 128, 32),
                    prep(P[9], 0, 128, 128, 128, 128),
                    prep(P[11],0, 128, 128, 128, 128) };
  const float* eiB[3] = { P[6], P[8], P[10] };
  float* noW[3] = { prep(P[84], 0, 128, 128, 128, 128),
                    prep(P[86], 0, 128, 128, 128, 128),
                    prep(P[88], 0, 128, 3, 16, 128) };
  const float* noB[3] = { P[83], P[85], P[87] };
  int eK[3] = {384,128,128}, nK[3] = {256,128,128};
  float *eW[6][3], *nW[6][3]; const float *eB[6][3], *nB[6][3];
  for (int L = 0; L < 6; ++L){
    int bi = (L < 3) ? (41 + L*12) : (89 + (L-3)*12);
    for (int j = 0; j < 3; ++j){
      eB[L][j] = P[bi + 2*j];
      eW[L][j] = prep(P[bi + 2*j + 1], 0, eK[j], 128, 128, eK[j]);
      nB[L][j] = P[bi + 6 + 2*j];
      nW[L][j] = prep(P[bi + 6 + 2*j + 1], 0, nK[j], 128, 128, nK[j]);
    }
  }
  float* Wt1a = prep(P[22], 0, 64, 32, 32, 64);    // in_kernel l0 query half
  float* Wt1b = prep(P[22], 64, 67, 32, 32, 96);   // in_kernel l0 data half (ye+f)
  float* Wt2i = prep(P[24], 0, 32, 64, 64, 32);
  float* Wt3i = prep(P[26], 0, 64, 3, 16, 64);
  float* Wl1  = prep(P[28], 0, 3, 32, 32, 32);
  float* Wl2  = prep(P[30], 0, 32, 32, 32, 32);
  float* Ws[2] = { prep(P[14], 0, 32, 32, 32, 32), prep(P[18], 0, 32, 32, 32, 32) };
  const float* Wsb[2] = { P[13], P[17] };
  const float* FNOwi[2] = { P[15], P[19] };
  const float* FNOwr[2] = { P[16], P[20] };
  float* WtqA2 = prep(P[32], 0, 64, 32, 32, 96);   // out_kernel l0 query (ye in yf cols 0..63)
  float* WtB2  = prep(P[32], 64, 64, 32, 32, 64);  // out_kernel l0 data (qe)
  float* Wt2o  = prep(P[34], 0, 32, 64, 64, 32);
  float* Wt3o  = prep(P[36], 0, 64, 32, 32, 64);
  float* Wp1   = prep(P[38], 0, 32, 16, 16, 32);
  float* Wp2   = prep(P[40], 0, 16, 128, 128, 32);

  // ---- activations ----
  float* h    = alloc((size_t)MN*128);
  float* e    = alloc((size_t)Ee*128);
  float* tA   = alloc((size_t)Ee*128);
  float* tB   = alloc((size_t)Ee*128);
  float* eC   = alloc((size_t)Ee*384);
  float* agg  = alloc((size_t)MN*128);
  float* nC   = alloc((size_t)MN*256);
  float* X0   = alloc((size_t)MN*64);
  float* E0   = alloc((size_t)Ee*32);
  float* out16= alloc((size_t)MN*16);
  float* p2d  = alloc((size_t)MN*2);
  float* yf   = alloc((size_t)MN*96);
  float* B1   = alloc((size_t)MN*32);
  float* qA2  = alloc((size_t)MN*32);
  float* zb   = alloc((size_t)MN*32);
  float* pb   = alloc((size_t)MN*32);
  float* w_in = alloc((size_t)256*MN);
  float* w_out= alloc((size_t)Nn*256);
  float* qe   = alloc(256*64);
  float* qA   = alloc(256*32);
  float* B2   = alloc(256*32);
  float* latP = alloc(256*32);
  float* L1   = alloc(256*32);
  float* u    = alloc(256*32);
  float* us   = alloc(256*32);
  float* spec = alloc(256*32);
  float* ufr  = alloc(4608); float* ufi = alloc(4608);
  float* ofr  = alloc(4608); float* ofi = alloc(4608);
  (void)ws_size; (void)in_sizes; (void)out_size;

  // ---- encoders ----
  k_node_in_build<<<MN, 64, 0, stream>>>(xv, pos, P[12], X0);
  gemm_go(stream, X0, niW[0], niB[0], tA, MN, MN, 64, 128, 128, 128, 1);
  gemm_go(stream, tA, niW[1], niB[1], tB, MN, MN, 128, 128, 128, 128, 1);
  gemm_go(stream, tB, niW[2], niB[2], tA, MN, MN, 128, 128, 128, 128, 0);
  k_layernorm<<<(MN+7)/8, 256, 0, stream>>>(tA, h, MN);
  k_edge_in_build<<<Ee, 32, 0, stream>>>(ea, E0);
  gemm_go(stream, E0, eiW[0], eiB[0], tA, Ee, Ee, 32, 128, 128, 128, 1);
  gemm_go(stream, tA, eiW[1], eiB[1], tB, Ee, Ee, 128, 128, 128, 128, 1);
  gemm_go(stream, tB, eiW[2], eiB[2], tA, Ee, Ee, 128, 128, 128, 128, 0);
  k_layernorm<<<(Ee+7)/8, 256, 0, stream>>>(tA, e, Ee);

  auto interaction = [&](int L){
    k_edge_concat<<<Ee, 128, 0, stream>>>(h, e, src, dst, eC);
    gemm_go(stream, eC, eW[L][0], eB[L][0], tA, Ee, Ee, 384, 128, 128, 128, 1);
    gemm_go(stream, tA, eW[L][1], eB[L][1], tB, Ee, Ee, 128, 128, 128, 128, 1);
    gemm_go(stream, tB, eW[L][2], eB[L][2], tA, Ee, Ee, 128, 128, 128, 128, 0);
    k_layernorm<<<(Ee+7)/8, 256, 0, stream>>>(tA, tB, Ee);
    k_scale_rows<<<Ee, 128, 0, stream>>>(tB, nd);
    k_add<<<1024, 256, 0, stream>>>(e, tB, (long)Ee*128);
    k_zero<<<256, 256, 0, stream>>>(agg, (long)MN*128);
    k_scatter_add<<<Ee, 128, 0, stream>>>(tB, dst, agg);
    k_node_concat<<<MN, 128, 0, stream>>>(h, agg, nC);
    gemm_go(stream, nC, nW[L][0], nB[L][0], tA, MN, MN, 256, 128, 128, 128, 1);
    gemm_go(stream, tA, nW[L][1], nB[L][1], tB, MN, MN, 128, 128, 128, 128, 1);
    gemm_go(stream, tB, nW[L][2], nB[L][2], tA, MN, MN, 128, 128, 128, 128, 0);
    k_layernorm<<<(MN+7)/8, 256, 0, stream>>>(tA, tB, MN);
    k_add<<<256, 256, 0, stream>>>(h, tB, (long)MN*128);
  };
  for (int L = 0; L < 3; ++L) interaction(L);

  // ---- node_out -> out (f for GINO) ----
  gemm_go(stream, h,  noW[0], noB[0], tA, MN, MN, 128, 128, 128, 128, 1);
  gemm_go(stream, tA, noW[1], noB[1], tB, MN, MN, 128, 128, 128, 128, 1);
  gemm_go(stream, tB, noW[2], noB[2], out16, MN, Nn, 128, 16, 3, 16, 0);

  // ---- GINO ----
  k_project2d<<<(Nn+127)/128, 128, 0, stream>>>(rp, P[126], P[125], p2d);
  k_qe<<<1, 256, 0, stream>>>(qe);
  k_zero<<<256, 256, 0, stream>>>(yf, (long)MN*96);
  k_yf<<<Nn, 64, 0, stream>>>(p2d, out16, yf);
  k_w_in<<<256, 256, 0, stream>>>(p2d, w_in);
  k_w_out<<<Nn, 256, 0, stream>>>(p2d, w_out);
  gemm_go(stream, qe, Wt1a, nullptr, qA, 256, 256, 64, 32, 32, 32, 0);
  gemm_go(stream, yf, Wt1b, P[21],   B1, MN, MN, 96, 32, 32, 32, 0);
  k_zero<<<32, 256, 0, stream>>>(latP, 256*32);
  k_gino_in<<<dim3(256, 10), 128, 0, stream>>>(qA, B1, Wt2i, P[23], Wt3i, P[25],
                                               w_in, latP, 8);
  gemm_go(stream, latP, Wl1, P[27], L1, 256, 256, 32, 32, 32, 32, 2);
  gemm_go(stream, L1,   Wl2, P[29], u,  256, 256, 32, 32, 32, 32, 0);
  for (int f = 0; f < 2; ++f){
    k_rfft2<<<36, 128, 0, stream>>>(u, ufr, ufi);
    k_specmul<<<36, 128, 0, stream>>>(ufr, ufi, FNOwr[f], FNOwi[f], ofr, ofi);
    k_irfft2<<<64, 128, 0, stream>>>(ofr, ofi, spec);
    gemm_go(stream, u, Ws[f], Wsb[f], us, 256, 256, 32, 32, 32, 32, 0);
    k_gelu_add<<<64, 128, 0, stream>>>(u, spec, us);
  }
  gemm_go(stream, yf, WtqA2, nullptr, qA2, MN, MN, 96, 32, 32, 32, 0);
  gemm_go(stream, qe, WtB2,  P[31],   B2, 256, 256, 64, 32, 32, 32, 0);
  k_gino_out<<<Nn/4, 128, 0, stream>>>(qA2, B2, Wt2o, P[33], Wt3o, P[35], w_out, u, zb);
  k_zero<<<256, 256, 0, stream>>>(pb, (long)MN*32);
  gemm_go(stream, zb, Wp1, P[37], pb, MN, MN, 32, 16, 16, 32, 2);
  gemm_go(stream, pb, Wp2, P[39], h,  MN, MN, 32, 128, 128, 128, 0);

  // ---- out interactions + final head ----
  for (int L = 3; L < 6; ++L) interaction(L);
  gemm_go(stream, h,  noW[0], noB[0], tA, MN, MN, 128, 128, 128, 128, 1);
  gemm_go(stream, tA, noW[1], noB[1], tB, MN, MN, 128, 128, 128, 128, 1);
  gemm_go(stream, tB, noW[2], noB[2], (float*)d_out, MN, Nn, 128, 16, 3, 3, 0);
}
